// ParallelMLPBase_11793980195161
// MI455X (gfx1250) — matrix operand order repaired and checked by reading, not compiled, not run on hardware
//
#include <hip/hip_runtime.h>
#include <hip/hip_bf16.h>
#include <cstdint>

// Problem constants (match reference)
#define N_TOK 8192
#define DDIM  2048
#define HDIM  2048
#define NEXP  8
#define TOPK  2
#define CAPAC 4096
#define NTCOP (N_TOK*TOPK)

// Tiling
#define MTILE 128
#define NTILE 128
#define KSTEP 64        // two 32-wide WMMA k-substeps per LDS stage
#define LDS_STRIDE 72   // padded halfs per LDS row (144B: 16B aligned, bank-spread)

typedef _Float16 h16;
typedef __attribute__((ext_vector_type(4)))  _Float16 v4h;
typedef __attribute__((ext_vector_type(8)))  _Float16 v8h;
typedef __attribute__((ext_vector_type(16))) _Float16 v16h;
typedef __attribute__((ext_vector_type(4)))  float    v4f;
typedef __attribute__((ext_vector_type(8)))  float    v8f;
typedef __attribute__((ext_vector_type(4)))  unsigned int v4u;
typedef __attribute__((ext_vector_type(4)))  int      v4i;
typedef __attribute__((ext_vector_type(8)))  int      v8i;

#ifndef __has_builtin
#define __has_builtin(x) 0
#endif

#if __has_builtin(__builtin_amdgcn_global_load_async_to_lds_b128)
#define USE_ASYNC_LDS 1
#else
#define USE_ASYNC_LDS 0
#endif

#if __has_builtin(__builtin_amdgcn_tensor_load_to_lds)
#define USE_TDM 1
#else
#define USE_TDM 0
#endif

// Typed AS casts via integers: LDS offset == low 32 bits of flat address
#define AS_GLB(p) ((__attribute__((address_space(1))) v4i*)(uintptr_t)(p))
#define AS_LDS(p) ((__attribute__((address_space(3))) v4i*)(unsigned)(uintptr_t)(p))

__device__ __forceinline__ void wait_async0() {
#if __has_builtin(__builtin_amdgcn_s_wait_asynccnt)
    __builtin_amdgcn_s_wait_asynccnt(0);
#else
    asm volatile("s_wait_asynccnt 0x0" ::: "memory");
#endif
}

__device__ __forceinline__ void wait_tensor0() {
#if __has_builtin(__builtin_amdgcn_s_wait_tensorcnt)
    __builtin_amdgcn_s_wait_tensorcnt(0);
#else
    asm volatile("s_wait_tensorcnt 0x0" ::: "memory");
#endif
}

// Issue one TDM 2D tile load: 128 rows x 64 f16, row padding 4 DW per 32 DW -> LDS stride 72 halfs
__device__ __forceinline__ void tdm_issue_128x64(h16* ldsbase, const h16* gsrc, unsigned stride_elems) {
#if USE_TDM
    unsigned lds_off = (unsigned)(uintptr_t)ldsbase;
    unsigned long long ga = (unsigned long long)(uintptr_t)gsrc;
    v4u g0;
    g0[0] = 1u;                                   // count=1 valid descriptor
    g0[1] = lds_off;                              // lds_addr
    g0[2] = (unsigned)(ga & 0xFFFFFFFFu);         // global_addr lo
    g0[3] = (unsigned)((ga >> 32) & 0x1FFFFFFu) | (2u << 30);  // addr hi | type=2
    v8i g1;
    g1[0] = (int)((1u << 16) | (1u << 20) | (4u << 22) | (3u << 25));
    //          data_size=2B   pad_en      interval=32dw  amount=4dw
    g1[1] = (int)(64u << 16);        // tensor_dim0 = 64
    g1[2] = (int)(128u << 16);       // tensor_dim1 = 128
    g1[3] = (int)(64u << 16);        // tile_dim0 = 64
    g1[4] = (int)(128u);             // tile_dim1 = 128, tile_dim2 = 0
    g1[5] = (int)stride_elems;       // tensor_dim0_stride (elements)
    g1[6] = 0;
    g1[7] = 0;
    v4i gz; gz[0] = 0; gz[1] = 0; gz[2] = 0; gz[3] = 0;
#if __clang_major__ >= 23
    v8i gz8;
#pragma unroll
    for (int j = 0; j < 8; ++j) gz8[j] = 0;
    __builtin_amdgcn_tensor_load_to_lds(g0, g1, gz, gz, gz8, 0);
#else
    __builtin_amdgcn_tensor_load_to_lds(g0, g1, gz, gz, 0);
#endif
#else
    (void)ldsbase; (void)gsrc; (void)stride_elems;
#endif
}

__device__ __forceinline__ v16h cat16(v8h lo, v8h hi) {
    return __builtin_shufflevector(lo, hi, 0,1,2,3,4,5,6,7,8,9,10,11,12,13,14,15);
}

// ---------------- routing ----------------

__global__ void moe_init_kernel(int* cnt) {
    if (threadIdx.x < NEXP) cnt[threadIdx.x] = 0;
}

__global__ void moe_route_kernel(const int* __restrict__ ei,
                                 int* __restrict__ cnt,
                                 int* __restrict__ row_tok,
                                 int* __restrict__ copy_dest) {
    int t = blockIdx.x * blockDim.x + threadIdx.x;
    if (t >= NTCOP) return;
    int e = ei[t];
    int slot = atomicAdd(&cnt[e], 1);
    row_tok[e * CAPAC + slot] = t >> 1;     // TOPK == 2
    copy_dest[t] = e * CAPAC + slot;
}

// ---------------- precision conversion / transpose passes ----------------

__global__ __launch_bounds__(256)
void convert_x_kernel(const float* __restrict__ x, h16* __restrict__ xh) {
    size_t i = ((size_t)blockIdx.x * blockDim.x + threadIdx.x) * 8;
    v4f a = *(const v4f*)(x + i);
    v4f b = *(const v4f*)(x + i + 4);
    v8h o;
#pragma unroll
    for (int j = 0; j < 4; ++j) { o[j] = (h16)a[j]; o[4 + j] = (h16)b[j]; }
    *(v8h*)(xh + i) = o;
}

__global__ void zero_row_kernel(h16* __restrict__ xh) {
    int i = threadIdx.x * 8;
    v8h z;
#pragma unroll
    for (int j = 0; j < 8; ++j) z[j] = (h16)0.0f;
    *(v8h*)(xh + (size_t)N_TOK * DDIM + i) = z;
}

// W (E, 2048 k, 2048 n) f32 -> Wt (E, 2048 n, 2048 k) f16
__global__ __launch_bounds__(256)
void transpose_w_kernel(const float* __restrict__ src, h16* __restrict__ dst) {
    __shared__ h16 tile[32][33];
    const int e  = blockIdx.z;
    const int k0 = blockIdx.x * 32;
    const int n0 = blockIdx.y * 32;
    const int tx = threadIdx.x & 31;
    const int ty = threadIdx.x >> 5;   // 0..7
    const float* s = src + (size_t)e * DDIM * HDIM;
    h16* d = dst + (size_t)e * DDIM * HDIM;
#pragma unroll
    for (int i = 0; i < 4; ++i) {
        int k = ty + 8 * i;
        tile[k][tx] = (h16)s[(size_t)(k0 + k) * 2048 + n0 + tx];
    }
    __syncthreads();
#pragma unroll
    for (int i = 0; i < 4; ++i) {
        int n = ty + 8 * i;
        d[(size_t)(n0 + n) * 2048 + k0 + tx] = tile[tx][n];
    }
}

// ---------------- GEMM1: Act = silu(X W1) * (X W3), f16 out ----------------

__global__ __launch_bounds__(256)
void moe_gemm1_kernel(const h16*  __restrict__ xh,
                      const h16*  __restrict__ wt1,   // (E, H n, D k) f16
                      const h16*  __restrict__ wt3,
                      const int*  __restrict__ bspe,
                      const int*  __restrict__ row_tok,
                      h16*        __restrict__ act) {
    const int e  = blockIdx.z;
    const int m0 = blockIdx.y * MTILE;
    const int h0 = blockIdx.x * NTILE;
    const int cntE = bspe[e];
    if (m0 >= cntE) return;

    __shared__ h16 A2 [2][MTILE][LDS_STRIDE];
    __shared__ h16 B1s[2][NTILE][LDS_STRIDE];
    __shared__ h16 B3s[2][NTILE][LDS_STRIDE];

    const int tid  = threadIdx.x;
    const int lane = tid & 31;
    const int wv   = tid >> 5;
    const int wm   = wv & 3;        // 4 waves in M
    const int wn   = wv >> 2;       // 2 waves in N
    const int lr   = lane & 15;
    const int lg   = lane >> 4;

    // A gather: thread owns (row = tid>>1, 32-col chunk)
    const int arow = tid >> 1;
    const int acq  = (tid & 1) * 32;
    int tok = N_TOK;  // zero row
    if (m0 + arow < cntE) tok = row_tok[e * CAPAC + m0 + arow];
    const h16* asrc = xh + (size_t)tok * DDIM + acq;
    h16* aldst[2] = { &A2[0][arow][acq], &A2[1][arow][acq] };

    // B tile origins (transposed f16 weights)
    const h16* w1tile = wt1 + ((size_t)e * HDIM + h0) * DDIM;
    const h16* w3tile = wt3 + ((size_t)e * HDIM + h0) * DDIM;
#if !USE_TDM
    const h16* w1row = w1tile + (size_t)arow * DDIM + acq;
    const h16* w3row = w3tile + (size_t)arow * DDIM + acq;
#endif

    v8f acc1[2][4];
    v8f acc3[2][4];
#pragma unroll
    for (int i = 0; i < 2; ++i)
#pragma unroll
        for (int j = 0; j < 4; ++j) { acc1[i][j] = (v8f)(0.0f); acc3[i][j] = (v8f)(0.0f); }

    // ---- stage issue helpers ----
    auto issue_stage = [&](int kk, int b) {
#if USE_ASYNC_LDS
        __builtin_amdgcn_global_load_async_to_lds_b128(AS_GLB(asrc + kk), AS_LDS(aldst[b]), 0, 0);
        __builtin_amdgcn_global_load_async_to_lds_b128(AS_GLB(asrc + kk), AS_LDS(aldst[b]), 16, 0);
        __builtin_amdgcn_global_load_async_to_lds_b128(AS_GLB(asrc + kk), AS_LDS(aldst[b]), 32, 0);
        __builtin_amdgcn_global_load_async_to_lds_b128(AS_GLB(asrc + kk), AS_LDS(aldst[b]), 48, 0);
#else
#pragma unroll
        for (int c = 0; c < 4; ++c)
            *(v8h*)(aldst[b] + 8 * c) = *(const v8h*)(asrc + kk + 8 * c);
#endif
#if USE_TDM
        if (wv == 0) {
            tdm_issue_128x64(&B1s[b][0][0], w1tile + kk, DDIM);
            tdm_issue_128x64(&B3s[b][0][0], w3tile + kk, DDIM);
        }
#else
#pragma unroll
        for (int c = 0; c < 4; ++c) {
            *(v8h*)(&B1s[b][arow][acq] + 8 * c) = *(const v8h*)(w1row + kk + 8 * c);
            *(v8h*)(&B3s[b][arow][acq] + 8 * c) = *(const v8h*)(w3row + kk + 8 * c);
        }
#endif
    };
    auto wait_stage = [&]() {
#if USE_ASYNC_LDS
        wait_async0();
#endif
#if USE_TDM
        if (wv == 0) wait_tensor0();
#endif
        __syncthreads();
    };

    // prologue: fill buffer 0
    issue_stage(0, 0);
    wait_stage();

    int p = 0;
    for (int k0 = 0; k0 < DDIM; k0 += KSTEP) {
        // issue next stage into other buffer (overlaps with compute below)
        if (k0 + KSTEP < DDIM) issue_stage(k0 + KSTEP, p ^ 1);

        // compute from buffer p: two 32-wide k-substeps
#pragma unroll
        for (int s = 0; s < 2; ++s) {
            v16h afr[2];
#pragma unroll
            for (int i = 0; i < 2; ++i) {
                int r = wm * 32 + i * 16 + lr;
                v8h a0 = *(const v8h*)&A2[p][r][s * 32 + lg * 16];
                v8h a1 = *(const v8h*)&A2[p][r][s * 32 + 8 + lg * 16];
                afr[i] = cat16(a0, a1);
            }
            {
                v16h bf[4];
#pragma unroll
                for (int nf = 0; nf < 4; ++nf) {
                    const h16* q = &B1s[p][wn * 64 + nf * 16 + lr][s * 32 + lg * 16];
                    bf[nf] = cat16(*(const v8h*)q, *(const v8h*)(q + 8));
                }
#pragma unroll
                for (int nf = 0; nf < 4; ++nf) {
                    acc1[0][nf] = __builtin_amdgcn_wmma_f32_16x16x32_f16(
                        false, afr[0], false, bf[nf], (short)0, acc1[0][nf], false, false);
                    acc1[1][nf] = __builtin_amdgcn_wmma_f32_16x16x32_f16(
                        false, afr[1], false, bf[nf], (short)0, acc1[1][nf], false, false);
                }
            }
            {
                v16h bf[4];
#pragma unroll
                for (int nf = 0; nf < 4; ++nf) {
                    const h16* q = &B3s[p][wn * 64 + nf * 16 + lr][s * 32 + lg * 16];
                    bf[nf] = cat16(*(const v8h*)q, *(const v8h*)(q + 8));
                }
#pragma unroll
                for (int nf = 0; nf < 4; ++nf) {
                    acc3[0][nf] = __builtin_amdgcn_wmma_f32_16x16x32_f16(
                        false, afr[0], false, bf[nf], (short)0, acc3[0][nf], false, false);
                    acc3[1][nf] = __builtin_amdgcn_wmma_f32_16x16x32_f16(
                        false, afr[1], false, bf[nf], (short)0, acc3[1][nf], false, false);
                }
            }
        }

        wait_stage();
        p ^= 1;
    }

    // ---- epilogue: SwiGLU, store f16 ----
#pragma unroll
    for (int i = 0; i < 2; ++i) {
#pragma unroll
        for (int nf = 0; nf < 4; ++nf) {
            int grow = m0 + wm * 32 + i * 16 + lg * 8;
            int gcol = h0 + wn * 64 + nf * 16 + lr;
            h16* dst = act + ((size_t)e * CAPAC + grow) * HDIM + gcol;
            v8f g = acc1[i][nf];
            v8f u = acc3[i][nf];
#pragma unroll
            for (int j = 0; j < 8; ++j) {
                float gv = g[j];
                float sv = gv / (1.0f + __expf(-gv));
                dst[(size_t)j * HDIM] = (h16)(sv * u[j]);
            }
        }
    }
}

// ---------------- GEMM2: Out = Act W2, f16 out ----------------

__global__ __launch_bounds__(256)
void moe_gemm2_kernel(const h16*  __restrict__ act,
                      const h16*  __restrict__ wt2,   // (E, D n, H k) f16
                      const int*  __restrict__ bspe,
                      h16*        __restrict__ outb) {
    const int e  = blockIdx.z;
    const int m0 = blockIdx.y * MTILE;
    const int d0 = blockIdx.x * NTILE;
    const int cntE = bspe[e];
    if (m0 >= cntE) return;

    __shared__ h16 A2 [2][MTILE][LDS_STRIDE];
    __shared__ h16 B2s[2][NTILE][LDS_STRIDE];

    const int tid  = threadIdx.x;
    const int lane = tid & 31;
    const int wv   = tid >> 5;
    const int wm   = wv & 3;
    const int wn   = wv >> 2;
    const int lr   = lane & 15;
    const int lg   = lane >> 4;

    const int arow = tid >> 1;
    const int acq  = (tid & 1) * 32;
    const h16* atile0 = act + ((size_t)e * CAPAC + m0) * HDIM;
    const h16* w2tile = wt2 + ((size_t)e * DDIM + d0) * HDIM;
#if !USE_TDM
    const h16* arowp = atile0 + (size_t)arow * HDIM + acq;
    const h16* w2row = w2tile + (size_t)arow * HDIM + acq;
    h16* aldst[2] = { &A2[0][arow][acq], &A2[1][arow][acq] };
#endif

    v8f acc[2][4];
#pragma unroll
    for (int i = 0; i < 2; ++i)
#pragma unroll
        for (int j = 0; j < 4; ++j) acc[i][j] = (v8f)(0.0f);

    auto issue_stage = [&](int kk, int b) {
#if USE_TDM
        if (wv == 0) {
            tdm_issue_128x64(&A2[b][0][0],  atile0 + kk, HDIM);
            tdm_issue_128x64(&B2s[b][0][0], w2tile + kk, HDIM);
        }
#elif USE_ASYNC_LDS
        __builtin_amdgcn_global_load_async_to_lds_b128(AS_GLB(arowp + kk), AS_LDS(aldst[b]), 0, 0);
        __builtin_amdgcn_global_load_async_to_lds_b128(AS_GLB(arowp + kk), AS_LDS(aldst[b]), 16, 0);
        __builtin_amdgcn_global_load_async_to_lds_b128(AS_GLB(arowp + kk), AS_LDS(aldst[b]), 32, 0);
        __builtin_amdgcn_global_load_async_to_lds_b128(AS_GLB(arowp + kk), AS_LDS(aldst[b]), 48, 0);
#pragma unroll
        for (int c = 0; c < 4; ++c)
            *(v8h*)(&B2s[b][arow][acq] + 8 * c) = *(const v8h*)(w2row + kk + 8 * c);
#else
#pragma unroll
        for (int c = 0; c < 4; ++c) {
            *(v8h*)(aldst[b] + 8 * c) = *(const v8h*)(arowp + kk + 8 * c);
            *(v8h*)(&B2s[b][arow][acq] + 8 * c) = *(const v8h*)(w2row + kk + 8 * c);
        }
#endif
    };
    auto wait_stage = [&]() {
#if USE_TDM
        if (wv == 0) wait_tensor0();
#elif USE_ASYNC_LDS
        wait_async0();
#endif
        __syncthreads();
    };

    issue_stage(0, 0);
    wait_stage();

    int p = 0;
    for (int k0 = 0; k0 < HDIM; k0 += KSTEP) {
        if (k0 + KSTEP < HDIM) issue_stage(k0 + KSTEP, p ^ 1);

#pragma unroll
        for (int s = 0; s < 2; ++s) {
            v16h afr[2];
#pragma unroll
            for (int i = 0; i < 2; ++i) {
                int r = wm * 32 + i * 16 + lr;
                v8h a0 = *(const v8h*)&A2[p][r][s * 32 + lg * 16];
                v8h a1 = *(const v8h*)&A2[p][r][s * 32 + 8 + lg * 16];
                afr[i] = cat16(a0, a1);
            }
            v16h bf[4];
#pragma unroll
            for (int nf = 0; nf < 4; ++nf) {
                const h16* q = &B2s[p][wn * 64 + nf * 16 + lr][s * 32 + lg * 16];
                bf[nf] = cat16(*(const v8h*)q, *(const v8h*)(q + 8));
            }
#pragma unroll
            for (int nf = 0; nf < 4; ++nf) {
                acc[0][nf] = __builtin_amdgcn_wmma_f32_16x16x32_f16(
                    false, afr[0], false, bf[nf], (short)0, acc[0][nf], false, false);
                acc[1][nf] = __builtin_amdgcn_wmma_f32_16x16x32_f16(
                    false, afr[1], false, bf[nf], (short)0, acc[1][nf], false, false);
            }
        }

        wait_stage();
        p ^= 1;
    }

#pragma unroll
    for (int i = 0; i < 2; ++i) {
#pragma unroll
        for (int nf = 0; nf < 4; ++nf) {
            int grow = m0 + wm * 32 + i * 16 + lg * 8;
            int gcol = d0 + wn * 64 + nf * 16 + lr;
            h16* dst = outb + ((size_t)e * CAPAC + grow) * DDIM + gcol;
            v8f c = acc[i][nf];
#pragma unroll
            for (int j = 0; j < 8; ++j)
                dst[(size_t)j * DDIM] = (h16)c[j];
        }
    }
}

// ---------------- combine: y[n] = sum_k w[n,k] * Out[dest[n,k]] ----------------

__global__ __launch_bounds__(256)
void moe_combine_kernel(const h16*  __restrict__ outb,
                        const float* __restrict__ ew,
                        const int*   __restrict__ copy_dest,
                        float*       __restrict__ y) {
    int n = blockIdx.x;
    int d = (blockIdx.y * blockDim.x + threadIdx.x) * 4;
    int t0 = n * TOPK, t1 = t0 + 1;
    int dst0 = copy_dest[t0];
    int dst1 = copy_dest[t1];
    float wa = ew[t0];
    float wb = ew[t1];
    v4h o0 = *(const v4h*)(outb + (size_t)dst0 * DDIM + d);
    v4h o1 = *(const v4h*)(outb + (size_t)dst1 * DDIM + d);
    v4f r;
#pragma unroll
    for (int j = 0; j < 4; ++j)
        r[j] = wa * (float)o0[j] + wb * (float)o1[j];
    *(v4f*)(y + (size_t)n * DDIM + d) = r;
}

// ---------------- launch ----------------

extern "C" void kernel_launch(void* const* d_in, const int* in_sizes, int n_in,
                              void* d_out, int out_size, void* d_ws, size_t ws_size,
                              hipStream_t stream) {
    const float* x    = (const float*)d_in[0];
    const float* ew   = (const float*)d_in[1];
    const float* w1   = (const float*)d_in[2];
    const float* w2   = (const float*)d_in[3];
    const float* w3   = (const float*)d_in[4];
    const int*   ei   = (const int*)d_in[5];
    const int*   bspe = (const int*)d_in[6];
    float* y = (float*)d_out;

    char* ws = (char*)d_ws;
    const size_t MB = 1ull << 20;
    int* cnt       = (int*)(ws);
    int* row_tok   = (int*)(ws + 1024);
    int* copy_dest = (int*)(ws + 1024 + (size_t)NEXP * CAPAC * 4);

    const size_t SZ_XH  = (size_t)(N_TOK + 1) * DDIM * 2;              // ~32MB
    const size_t SZ_WT  = (size_t)NEXP * DDIM * HDIM * 2;              // 64MB each
    const size_t SZ_ACT = (size_t)NEXP * CAPAC * HDIM * 2;             // 128MB

    size_t off = MB;
    h16* xh  = (h16*)(ws + off); off += (SZ_XH + MB - 1) & ~(MB - 1);
    h16* wt1 = (h16*)(ws + off); off += SZ_WT;
    h16* wt3 = (h16*)(ws + off); off += SZ_WT;
    h16* wt2 = (h16*)(ws + off); off += SZ_WT;
    h16* act = (h16*)(ws + off); off += SZ_ACT;
    h16* outb= (h16*)(ws + off);

    moe_init_kernel<<<dim3(1), dim3(64), 0, stream>>>(cnt);
    moe_route_kernel<<<dim3(NTCOP / 256), dim3(256), 0, stream>>>(ei, cnt, row_tok, copy_dest);

    convert_x_kernel<<<dim3((N_TOK * (DDIM / 8)) / 256), dim3(256), 0, stream>>>(x, xh);
    zero_row_kernel<<<dim3(1), dim3(256), 0, stream>>>(xh);
    transpose_w_kernel<<<dim3(DDIM / 32, HDIM / 32, NEXP), dim3(256), 0, stream>>>(w1, wt1);
    transpose_w_kernel<<<dim3(DDIM / 32, HDIM / 32, NEXP), dim3(256), 0, stream>>>(w3, wt3);
    transpose_w_kernel<<<dim3(HDIM / 32, DDIM / 32, NEXP), dim3(256), 0, stream>>>(w2, wt2);

    moe_gemm1_kernel<<<dim3(HDIM / NTILE, CAPAC / MTILE, NEXP), dim3(256), 0, stream>>>(
        xh, wt1, wt3, bspe, row_tok, act);
    moe_gemm2_kernel<<<dim3(DDIM / NTILE, CAPAC / MTILE, NEXP), dim3(256), 0, stream>>>(
        act, wt2, bspe, outb);
    moe_combine_kernel<<<dim3(N_TOK, DDIM / 1024), dim3(256), 0, stream>>>(
        outb, ew, copy_dest, y);
}
